// StateSpaceModelAttentionWithSSD_21088289423859
// MI455X (gfx1250) — matrix operand order repaired and checked by reading, not compiled, not run on hardware
//
#include <hip/hip_runtime.h>
#include <hip/hip_bf16.h>
#include <math.h>

// ---------------- problem constants ----------------
#define B_      8
#define S_      2048
#define STATE_  1024
#define HEADS_  8
#define HDIM_   128
#define NBLK_   8
#define MROWS_  (B_*S_)          // 16384

// ---------------- vector types (trivial ext-vectors, union-safe) ----------------
typedef __attribute__((ext_vector_type(16))) __bf16  v16bf;
typedef __attribute__((ext_vector_type(8)))  float   v8f;
typedef __attribute__((ext_vector_type(4)))  unsigned int u32x4;
typedef __attribute__((ext_vector_type(2)))  unsigned int u32x2;
typedef __attribute__((ext_vector_type(4)))  float   f32x4;

// ---------------- WMMA helper ----------------
__device__ inline v8f wmma_bf16(v16bf a, v16bf b, v8f c) {
    return __builtin_amdgcn_wmma_f32_16x16x32_bf16(false, a, false, b,
                                                   (short)0, c, false, false);
}

// B-matrix (32x16 bf16): e<8: K=8h+e; e>=8: K=16+8h+(e-8) (h=lane/16), matching A fragment
__device__ __host__ inline int klocB(int lane, int e) {
    return ((lane & 16) >> 1) + (e & 7) + ((e & 8) << 1);
}

// Load A fragment from an LDS tile (bf16, row-major, pitch in halfs, pitch%8==0)
// A-matrix (16x32 bf16) per ISA 7.12.2: VGPR0-3: K=0..7 (lanes0-15)/K=8..15 (lanes16-31);
// VGPR4-7: K=16..23 / 24..31 -> two contiguous b128 chunks per lane.
__device__ inline v16bf ld_fragA_lds(const __bf16* lds, int pitch, int row0, int kb, int lane) {
    int r   = row0 + (lane & 15);
    int hi8 = (lane & 16) >> 1;                 // 0 or 8
    const __bf16* p = lds + r * pitch + kb + hi8;
    union { v16bf v; u32x4 q[2]; } u;
    u.q[0] = *(const u32x4*)p;                  // K kb+hi .. +7
    u.q[1] = *(const u32x4*)(p + 16);           // K kb+16+hi .. +23
    return u.v;
}

// Load pre-fragmented B from global: [tile][lane][16 halfs] contiguous
__device__ inline v16bf ld_fragB_g(const __bf16* base, int tile, int lane) {
    const __bf16* p = base + ((size_t)tile * 32 + lane) * 16;
    union { v16bf v; u32x4 q[2]; } u;
    u.q[0] = *(const u32x4*)p;
    u.q[1] = *(const u32x4*)(p + 8);
    return u.v;
}

// ================= weight fragmentation =================
// B[k][n] = W[h][n][k]  (q = x @ W^T per head)
__global__ void frag_qkv_k(const float* Wq, const float* Wk, const float* Wv, __bf16* dst) {
    int g = blockIdx.x * blockDim.x + threadIdx.x;
    int lane = g & 31, tl = g >> 5;
    if (tl >= 3 * HEADS_ * 32) return;
    int mat = tl / (HEADS_ * 32);
    int rem = tl % (HEADS_ * 32);
    int h = rem >> 5, t = rem & 31;
    int ntile = t >> 2, kt = t & 3;
    const float* W = (mat == 0) ? Wq : ((mat == 1) ? Wk : Wv);
    const float* Wh = W + (size_t)h * HDIM_ * HDIM_;
    __bf16* o = dst + ((size_t)tl * 32 + lane) * 16;
    int n = ntile * 16 + (lane & 15);
    #pragma unroll
    for (int e = 0; e < 16; e++) {
        int k = kt * 32 + klocB(lane, e);
        o[e] = (__bf16)Wh[n * HDIM_ + k];       // W[v=n][d=k]
    }
}

// diagonal blocks of SSM matrices: B[k][n] = M[nb*128+k][nb*128+n]
__global__ void frag_abc_k(const float* A, const float* Bm, const float* C, __bf16* dst) {
    int g = blockIdx.x * blockDim.x + threadIdx.x;
    int lane = g & 31, tl = g >> 5;
    if (tl >= 3 * NBLK_ * 32) return;
    int mat = tl / (NBLK_ * 32);
    int rem = tl % (NBLK_ * 32);
    int nb = rem >> 5, t = rem & 31;
    int ntile = t >> 2, kt = t & 3;
    const float* Mx = (mat == 0) ? A : ((mat == 1) ? Bm : C);
    __bf16* o = dst + ((size_t)tl * 32 + lane) * 16;
    int n = nb * 128 + ntile * 16 + (lane & 15);
    #pragma unroll
    for (int e = 0; e < 16; e++) {
        int k = nb * 128 + kt * 32 + klocB(lane, e);
        o[e] = (__bf16)Mx[(size_t)k * STATE_ + n];
    }
}

// W_o row-major [k][n] direct
__global__ void frag_wo_k(const float* Wo, __bf16* dst) {
    int g = blockIdx.x * blockDim.x + threadIdx.x;
    int lane = g & 31, tl = g >> 5;
    if (tl >= 64 * 32) return;
    int ntile = tl >> 5, kt = tl & 31;
    __bf16* o = dst + ((size_t)tl * 32 + lane) * 16;
    int n = ntile * 16 + (lane & 15);
    #pragma unroll
    for (int e = 0; e < 16; e++) {
        int k = kt * 32 + klocB(lane, e);
        o[e] = (__bf16)Wo[(size_t)k * STATE_ + n];
    }
}

// ================= QKV projection (bf16 WMMA, 4-way interleaved tiles) =================
__global__ void __launch_bounds__(256)
qkv_k(const float* __restrict__ x, const __bf16* __restrict__ wfrag,
      __bf16* __restrict__ qo, __bf16* __restrict__ ko, __bf16* __restrict__ vo) {
    __shared__ __bf16 xs[64 * 136];
    int m0 = blockIdx.x * 64;
    int h  = blockIdx.y;
    int tid = threadIdx.x;
    {   // stage x tile (64 rows x 128 head-cols) fp32 -> bf16
        int row = tid >> 2, c0 = (tid & 3) * 32;
        const float* xp = x + (size_t)(m0 + row) * STATE_ + h * HDIM_ + c0;
        __bf16* sp = xs + row * 136 + c0;
        #pragma unroll
        for (int i = 0; i < 32; i += 8) {
            f32x4 f0 = *(const f32x4*)(xp + i);
            f32x4 f1 = *(const f32x4*)(xp + i + 4);
            union { u32x4 u; __bf16 b[8]; } o;
            o.b[0] = (__bf16)f0.x; o.b[1] = (__bf16)f0.y; o.b[2] = (__bf16)f0.z; o.b[3] = (__bf16)f0.w;
            o.b[4] = (__bf16)f1.x; o.b[5] = (__bf16)f1.y; o.b[6] = (__bf16)f1.z; o.b[7] = (__bf16)f1.w;
            *(u32x4*)(sp + i) = o.u;
        }
    }
    __syncthreads();
    int wave = tid >> 5, lane = tid & 31;
    // 96 tiles (mat,mtile,ntile); wave gets 12 = 3 quads; each quad shares (mat,mtile)
    for (int gq = 0; gq < 3; gq++) {
        int base = wave * 12 + gq * 4;
        int mat = base >> 5;
        int r = base & 31;
        int mtile = r >> 3, nt0 = r & 7;           // nt0 in {0,4}
        v8f acc[4];
        #pragma unroll
        for (int j = 0; j < 4; j++) acc[j] = (v8f){};
        #pragma unroll
        for (int kt = 0; kt < 4; kt++) {
            v16bf a = ld_fragA_lds(xs, 136, mtile * 16, kt * 32, lane);
            #pragma unroll
            for (int j = 0; j < 4; j++) {          // 4 independent WMMA chains
                v16bf b = ld_fragB_g(wfrag, (mat * HEADS_ + h) * 32 + (nt0 + j) * 4 + kt, lane);
                acc[j] = wmma_bf16(a, b, acc[j]);
            }
        }
        __bf16* outp = (mat == 0) ? qo : ((mat == 1) ? ko : vo);
        int rowb = m0 + mtile * 16 + ((lane >> 4) << 3);
        #pragma unroll
        for (int j = 0; j < 4; j++) {
            int col = h * HDIM_ + (nt0 + j) * 16 + (lane & 15);
            #pragma unroll
            for (int e = 0; e < 8; e++)
                outp[(size_t)(rowb + e) * STATE_ + col] = (__bf16)acc[j][e];
        }
    }
}

// ================= score + block-sum reduction =================
__global__ void __launch_bounds__(256)
score_k(const __bf16* __restrict__ q, const __bf16* __restrict__ k, const __bf16* __restrict__ v,
        float* __restrict__ sq, float* __restrict__ sk, float* __restrict__ sv,
        float* __restrict__ score_sum) {
    int row  = blockIdx.x * 8 + (threadIdx.x >> 5);
    int lane = threadIdx.x & 31;
    const __bf16* qr = q + (size_t)row * STATE_;
    const __bf16* kr = k + (size_t)row * STATE_;
    const __bf16* vr = v + (size_t)row * STATE_;
    float sc = 0.f;
    for (int h = 0; h < HEADS_; h++) {
        int c = h * 128 + lane * 4;
        union { u32x2 u; __bf16 b[4]; } uq, uk, uv;
        uq.u = *(const u32x2*)(qr + c);
        uk.u = *(const u32x2*)(kr + c);
        uv.u = *(const u32x2*)(vr + c);
        float pq = 0.f, pk = 0.f, pv = 0.f;
        #pragma unroll
        for (int i = 0; i < 4; i++) {
            float fq = (float)uq.b[i], fk = (float)uk.b[i], fv = (float)uv.b[i];
            pq += fq; pk += fk; pv += fv; sc += fq * fk;
        }
        #pragma unroll
        for (int off = 16; off; off >>= 1) {
            pq += __shfl_xor(pq, off, 32);
            pk += __shfl_xor(pk, off, 32);
            pv += __shfl_xor(pv, off, 32);
        }
        if (lane == 0) {
            sq[row * 8 + h] = pq; sk[row * 8 + h] = pk; sv[row * 8 + h] = pv;
        }
    }
    #pragma unroll
    for (int off = 16; off; off >>= 1) sc += __shfl_xor(sc, off, 32);
    if (lane == 0) atomicAdd(&score_sum[row / S_], sc);
}

// ================= gate: dwc, w_mean, per-batch a[b] =================
__global__ void dwc_k(const float* __restrict__ score_sum, float* __restrict__ acoef) {
    int lane = threadIdx.x;
    float dwc = 0.f;
    if (lane < B_) {
        float mean = score_sum[lane] * (1.f / (float)S_);
        float z = 10.f * mean / logf((float)S_ + 1.f);
        dwc = 1.f / (1.f + expf(-z));
    }
    float s = dwc;
    #pragma unroll
    for (int off = 16; off; off >>= 1) s += __shfl_xor(s, off, 32);
    float wmean = s * (1.f / (float)B_);
    if (lane < B_) {
        float a = (wmean > 0.99f) ? 1.f : ((wmean < 0.01f) ? 0.f : dwc);
        acoef[lane] = a;
    }
}

// ================= block-diagonal SSM (scaled-input trick) + residual =================
__global__ void __launch_bounds__(256)
ssm_k(const __bf16* __restrict__ q, const __bf16* __restrict__ k, const __bf16* __restrict__ v,
      const float* __restrict__ sq, const float* __restrict__ sk, const float* __restrict__ sv,
      const float* __restrict__ acoef, const __bf16* __restrict__ abcfrag,
      const float* __restrict__ x, float* __restrict__ ssm) {
    __shared__ __bf16 qs[64 * 136];
    __shared__ __bf16 ks[64 * 136];
    __shared__ __bf16 vs[64 * 136];
    int m0 = blockIdx.x * 64;                      // tile stays inside one batch (64 | 2048)
    float a = acoef[m0 / S_];
    int tid = threadIdx.x, wave = tid >> 5, lane = tid & 31;
    int mtile = wave >> 1;                          // wave's quad: same mtile, 4 ntiles
    int nt0   = (wave & 1) * 4;
    for (int nb = 0; nb < NBLK_; nb++) {
        __syncthreads();
        {   // stage scaled q/k/v block (coef = a*0.1*blocksum + (1-a))
            int row = tid >> 2, c0 = (tid & 3) * 32;
            int grow = m0 + row;
            float cq = a * 0.1f * sq[grow * 8 + nb] + (1.f - a);
            float ck = a * 0.1f * sk[grow * 8 + nb] + (1.f - a);
            float cv = a * 0.1f * sv[grow * 8 + nb] + (1.f - a);
            size_t goff = (size_t)grow * STATE_ + nb * 128 + c0;
            int loff = row * 136 + c0;
            #pragma unroll
            for (int i = 0; i < 32; i += 8) {
                union { u32x4 u; __bf16 b[8]; } tq, tk, tv, oq, ok, ov;
                tq.u = *(const u32x4*)(q + goff + i);
                tk.u = *(const u32x4*)(k + goff + i);
                tv.u = *(const u32x4*)(v + goff + i);
                #pragma unroll
                for (int j = 0; j < 8; j++) {
                    oq.b[j] = (__bf16)((float)tq.b[j] * cq);
                    ok.b[j] = (__bf16)((float)tk.b[j] * ck);
                    ov.b[j] = (__bf16)((float)tv.b[j] * cv);
                }
                *(u32x4*)(qs + loff + i) = oq.u;
                *(u32x4*)(ks + loff + i) = ok.u;
                *(u32x4*)(vs + loff + i) = ov.u;
            }
        }
        __syncthreads();
        v8f acc[4];
        #pragma unroll
        for (int j = 0; j < 4; j++) acc[j] = (v8f){};
        #pragma unroll
        for (int kt = 0; kt < 4; kt++) {
            const __bf16* mats[3] = { qs, ks, vs };
            #pragma unroll
            for (int m = 0; m < 3; m++) {
                v16bf aA = ld_fragA_lds(mats[m], 136, mtile * 16, kt * 32, lane);
                #pragma unroll
                for (int j = 0; j < 4; j++) {      // 4 independent chains per A frag
                    v16bf bB = ld_fragB_g(abcfrag,
                                          (m * NBLK_ + nb) * 32 + (nt0 + j) * 4 + kt, lane);
                    acc[j] = wmma_bf16(aA, bB, acc[j]);
                }
            }
        }
        int rowb = m0 + mtile * 16 + ((lane >> 4) << 3);
        #pragma unroll
        for (int j = 0; j < 4; j++) {
            int col = nb * 128 + (nt0 + j) * 16 + (lane & 15);
            #pragma unroll
            for (int e = 0; e < 8; e++) {
                size_t idx = (size_t)(rowb + e) * STATE_ + col;
                ssm[idx] = acc[j][e] + x[idx];     // residual
            }
        }
    }
}

// ================= LayerNorm (wave per row) =================
__global__ void __launch_bounds__(256)
ln_k(const float* __restrict__ ssm, const float* __restrict__ gamma,
     const float* __restrict__ beta, __bf16* __restrict__ normed) {
    int row  = blockIdx.x * 8 + (threadIdx.x >> 5);
    int lane = threadIdx.x & 31;
    const float* rp = ssm + (size_t)row * STATE_ + lane * 32;
    float vals[32];
    float s = 0.f, s2 = 0.f;
    #pragma unroll
    for (int i = 0; i < 32; i += 4) {
        f32x4 f = *(const f32x4*)(rp + i);
        vals[i] = f.x; vals[i + 1] = f.y; vals[i + 2] = f.z; vals[i + 3] = f.w;
        s  += f.x + f.y + f.z + f.w;
        s2 += f.x * f.x + f.y * f.y + f.z * f.z + f.w * f.w;
    }
    #pragma unroll
    for (int off = 16; off; off >>= 1) {
        s  += __shfl_xor(s,  off, 32);
        s2 += __shfl_xor(s2, off, 32);
    }
    float mu   = s  * (1.f / (float)STATE_);
    float var  = s2 * (1.f / (float)STATE_) - mu * mu;
    float rstd = rsqrtf(var + 1e-5f);
    const float* g  = gamma + lane * 32;
    const float* bt = beta  + lane * 32;
    __bf16* op = normed + (size_t)row * STATE_ + lane * 32;
    #pragma unroll
    for (int i = 0; i < 32; i += 8) {
        union { u32x4 u; __bf16 b[8]; } o;
        #pragma unroll
        for (int j = 0; j < 8; j++)
            o.b[j] = (__bf16)((vals[i + j] - mu) * rstd * g[i + j] + bt[i + j]);
        *(u32x4*)(op + i) = o.u;
    }
}

// ================= output GEMM: normed(bf16) @ W_o, async-to-LDS staging =================
__global__ void __launch_bounds__(256)
gemm_k(const __bf16* __restrict__ normed, const __bf16* __restrict__ wofrag,
       float* __restrict__ out) {
    __shared__ __bf16 as[128 * 40];
    int m0  = blockIdx.x * 128;
    int n0t = blockIdx.y * 8;                      // ntile base (128 output cols)
    int tid = threadIdx.x, wave = tid >> 5, lane = tid & 31;
    v8f acc[8];
    #pragma unroll
    for (int nt = 0; nt < 8; nt++) acc[nt] = (v8f){};
    int srow = tid >> 1, sc0 = (tid & 1) * 16;     // this thread's staging slot
    unsigned lds0 = (unsigned)(size_t)(as + srow * 40 + sc0);
    for (int kt = 0; kt < 32; kt++) {
        __syncthreads();
        {   // async DMA: 128 rows x 32 K halfs, 2 x b128 per thread, ASYNCcnt-tracked
            unsigned voff = (unsigned)(((size_t)(m0 + srow) * STATE_ + kt * 32 + sc0) * 2);
            asm volatile("global_load_async_to_lds_b128 %0, %1, %2"
                         :: "v"(lds0), "v"(voff), "s"(normed) : "memory");
            asm volatile("global_load_async_to_lds_b128 %0, %1, %2"
                         :: "v"(lds0 + 16u), "v"(voff + 16u), "s"(normed) : "memory");
            asm volatile("s_wait_asynccnt 0x0" ::: "memory");
        }
        __syncthreads();
        v16bf aA = ld_fragA_lds(as, 40, wave * 16, 0, lane);
        #pragma unroll
        for (int nt = 0; nt < 8; nt++) {           // 8 independent WMMA chains
            v16bf bB = ld_fragB_g(wofrag, (n0t + nt) * 32 + kt, lane);
            acc[nt] = wmma_bf16(aA, bB, acc[nt]);
        }
    }
    #pragma unroll
    for (int nt = 0; nt < 8; nt++) {
        int col  = (n0t + nt) * 16 + (lane & 15);
        int rowb = m0 + wave * 16 + ((lane >> 4) << 3);
        #pragma unroll
        for (int e = 0; e < 8; e++)
            out[(size_t)(rowb + e) * STATE_ + col] = acc[nt][e];
    }
}

// ================= host launch =================
extern "C" void kernel_launch(void* const* d_in, const int* in_sizes, int n_in,
                              void* d_out, int out_size, void* d_ws, size_t ws_size,
                              hipStream_t stream) {
    const float* x  = (const float*)d_in[0];
    const float* Wq = (const float*)d_in[1];
    const float* Wk = (const float*)d_in[2];
    const float* Wv = (const float*)d_in[3];
    const float* Wo = (const float*)d_in[4];
    const float* SA = (const float*)d_in[5];
    const float* SB = (const float*)d_in[6];
    const float* SC = (const float*)d_in[7];
    const float* gamma = (const float*)d_in[8];
    const float* beta  = (const float*)d_in[9];
    char* ws = (char*)d_ws;

    constexpr size_t SZ_WQKV = 768ull  * 32 * 16 * 2;   // 786,432
    constexpr size_t SZ_ABC  = 768ull  * 32 * 16 * 2;   // 786,432
    constexpr size_t SZ_WO   = 2048ull * 32 * 16 * 2;   // 2,097,152
    constexpr size_t SZ_SUMS = (size_t)MROWS_ * 8 * 4;  // 524,288
    constexpr size_t SZ_QKV1 = (size_t)MROWS_ * STATE_ * 2; // 33,554,432

    size_t o_wqkv  = 0;
    size_t o_abc   = o_wqkv + SZ_WQKV;
    size_t o_wo    = o_abc  + SZ_ABC;
    size_t o_score = o_wo   + SZ_WO;
    size_t o_acoef = o_score + 256;
    size_t o_sq    = o_acoef + 256;
    size_t o_sk    = o_sq + SZ_SUMS;
    size_t o_sv    = o_sk + SZ_SUMS;
    size_t o_q     = o_sv + SZ_SUMS;
    size_t o_k     = o_q + SZ_QKV1;
    size_t o_v     = o_k + SZ_QKV1;
    size_t o_ssm   = o_v + SZ_QKV1;
    size_t o_norm  = o_q;   // alias: q dead after ssm_k

    __bf16* wqkv = (__bf16*)(ws + o_wqkv);
    __bf16* abc  = (__bf16*)(ws + o_abc);
    __bf16* wof  = (__bf16*)(ws + o_wo);
    float*  scr  = (float*)(ws + o_score);
    float*  aco  = (float*)(ws + o_acoef);
    float*  sq   = (float*)(ws + o_sq);
    float*  sk   = (float*)(ws + o_sk);
    float*  sv   = (float*)(ws + o_sv);
    __bf16* qb   = (__bf16*)(ws + o_q);
    __bf16* kb   = (__bf16*)(ws + o_k);
    __bf16* vb   = (__bf16*)(ws + o_v);
    float*  ssmb = (float*)(ws + o_ssm);
    __bf16* nrm  = (__bf16*)(ws + o_norm);

    hipMemsetAsync(scr, 0, 256, stream);

    frag_qkv_k<<<96,  256, 0, stream>>>(Wq, Wk, Wv, wqkv);
    frag_abc_k<<<96,  256, 0, stream>>>(SA, SB, SC, abc);
    frag_wo_k <<<256, 256, 0, stream>>>(Wo, wof);

    qkv_k<<<dim3(MROWS_ / 64, HEADS_), 256, 0, stream>>>(x, wqkv, qb, kb, vb);
    score_k<<<MROWS_ / 8, 256, 0, stream>>>(qb, kb, vb, sq, sk, sv, scr);
    dwc_k<<<1, 32, 0, stream>>>(scr, aco);
    ssm_k<<<MROWS_ / 64, 256, 0, stream>>>(qb, kb, vb, sq, sk, sv, aco, abc, x, ssmb);
    ln_k<<<MROWS_ / 8, 256, 0, stream>>>(ssmb, gamma, beta, nrm);
    gemm_k<<<dim3(MROWS_ / 128, STATE_ / 128), 256, 0, stream>>>(nrm, wof, (float*)d_out);
}
